// BiRealConv2d_75170517614965
// MI455X (gfx1250) — 8-bit matrix operand order repaired and checked by reading, compiled, not run on hardware
//
#include <hip/hip_runtime.h>
#include <stdint.h>

typedef __attribute__((ext_vector_type(8))) int v8i;

#define N_    32
#define C_    128
#define K_    128
#define H_    112
#define W_    112
#define HW_   (H_ * W_)
#define KRED  1152
#define KSTEPS 18
#define LDSS  1168

__global__ void prep_x_kernel(const float* __restrict__ x,
                              uint32_t* __restrict__ xb4, int n4) {
  int i = blockIdx.x * blockDim.x + threadIdx.x;
  if (i >= n4) return;
  float4 v = ((const float4*)x)[i];
  uint32_t b0 = (uint8_t)(v.x > 0.f ? 1 : (v.x < 0.f ? -1 : 0));
  uint32_t b1 = (uint8_t)(v.y > 0.f ? 1 : (v.y < 0.f ? -1 : 0));
  uint32_t b2 = (uint8_t)(v.z > 0.f ? 1 : (v.z < 0.f ? -1 : 0));
  uint32_t b3 = (uint8_t)(v.w > 0.f ? 1 : (v.w < 0.f ? -1 : 0));
  xb4[i] = b0 | (b1 << 8) | (b2 << 16) | (b3 << 24);
}

__global__ void prep_w_scale_kernel(const float* __restrict__ w,
                                    float* __restrict__ scale) {
  __shared__ float red[64];
  int k = blockIdx.x;
  float s = 0.f;
  for (int r = threadIdx.x; r < KRED; r += 64)
    s += fabsf(w[(size_t)k * KRED + r]);
  red[threadIdx.x] = s;
  __syncthreads();
  for (int off = 32; off > 0; off >>= 1) {
    if ((int)threadIdx.x < off) red[threadIdx.x] += red[threadIdx.x + off];
    __syncthreads();
  }
  if (threadIdx.x == 0) scale[k] = red[0] / (float)KRED;
}

__global__ void prep_w_pack_kernel(const float* __restrict__ w,
                                   uint32_t* __restrict__ apack) {
  int gid = blockIdx.x * blockDim.x + threadIdx.x;
  if (gid >= 8 * KSTEPS * 32 * 8) return;
  int v    = gid & 7;
  int t    = gid >> 3;
  int lane = t & 31;  t >>= 5;
  int kk    = t % KSTEPS;
  int mtile = t / KSTEPS;
  int hh = lane >> 4;
  int m  = lane & 15;
  int kout = mtile * 16 + m;
  int koff = 32 * (v >> 2) + 16 * hh + 4 * (v & 3);
  uint32_t pack = 0;
#pragma unroll
  for (int b = 0; b < 4; ++b) {
    int kr = kk * 64 + koff + b;
    float wv = w[(size_t)kout * KRED + kr];
    uint32_t sb = (uint8_t)(wv > 0.f ? 1 : (wv < 0.f ? -1 : 0));
    pack |= sb << (8 * b);
  }
  apack[gid] = pack;
}

__global__ __launch_bounds__(256)
void bireal_gemm_kernel(const int8_t* __restrict__ xb,
                        const v8i* __restrict__ apack,
                        const float* __restrict__ scale,
                        float* __restrict__ out) {
  __shared__ __attribute__((aligned(16))) uint8_t ldsB[16 * LDSS];

  int blk = blockIdx.x;
  int wt  = blk % 7;
  int t   = blk / 7;
  int h   = t % H_;
  int n   = t / H_;
  int w0  = wt * 16;
  int tid = threadIdx.x;

  {
    int4* z = (int4*)ldsB;
#pragma unroll 2
    for (int i = tid; i < (16 * LDSS) / 16; i += 256) z[i] = int4{0, 0, 0, 0};
  }
  __syncthreads();

  const int8_t* xbn = xb + (size_t)n * C_ * HW_;
  for (int r = tid; r < KRED; r += 256) {
    int c  = r / 9;
    int tt = r - c * 9;
    int kh = tt / 3;
    int kw = tt - kh * 3;
    int hy = h + kh - 1;
    if (hy < 0 || hy >= H_) continue;
    const int8_t* src = xbn + ((size_t)c * H_ + hy) * W_;
    int wbase = w0 + kw - 1;
#pragma unroll
    for (int j = 0; j < 16; ++j) {
      int wx = wbase + j;
      if (wx >= 0 && wx < W_) {
        uint32_t laddr = (uint32_t)(uintptr_t)&ldsB[j * LDSS + r];
        asm volatile("global_load_async_to_lds_b8 %0, %1, off"
                     :: "v"(laddr), "v"(src + wx) : "memory");
      }
    }
  }
  asm volatile("s_wait_asynccnt 0x0" ::: "memory");
  __syncthreads();

  int wave = tid >> 5;
  int lane = tid & 31;
  int j    = lane & 15;
  int hh   = lane >> 4;

  const v8i* arow = apack + (size_t)wave * KSTEPS * 32 + lane;
  v8i acc = {};

#pragma unroll
  for (int kk = 0; kk < KSTEPS; ++kk) {
    v8i afrag = arow[(size_t)kk * 32];

    const uint8_t* bp = &ldsB[j * LDSS + kk * 64 + hh * 16];
    int4 b0 = *(const int4*)(bp);
    int4 b1 = *(const int4*)(bp + 32);
    v8i bfrag;
    bfrag[0] = b0.x; bfrag[1] = b0.y; bfrag[2] = b0.z; bfrag[3] = b0.w;
    bfrag[4] = b1.x; bfrag[5] = b1.y; bfrag[6] = b1.z; bfrag[7] = b1.w;

    acc = __builtin_amdgcn_wmma_i32_16x16x64_iu8(
         true, afrag,  true, bfrag, acc,
         false,  false);
  }

  float* ob = out + (size_t)n * K_ * HW_ + (size_t)h * W_ + w0 + j;
#pragma unroll
  for (int r = 0; r < 8; ++r) {
    int kout = wave * 16 + hh * 8 + r;
    ob[(size_t)kout * HW_] = scale[kout] * (float)acc[r];
  }
}

extern "C" void kernel_launch(void* const* d_in, const int* in_sizes, int n_in,
                              void* d_out, int out_size, void* d_ws, size_t ws_size,
                              hipStream_t stream) {
  const float* x = (const float*)d_in[0];
  const float* w = (const float*)d_in[1];
  float* out = (float*)d_out;

  int8_t* xb = (int8_t*)d_ws;
  const size_t XB_BYTES = (size_t)N_ * C_ * HW_;
  uint32_t* apack = (uint32_t*)(xb + XB_BYTES);
  const size_t APACK_DW = (size_t)8 * KSTEPS * 32 * 8;
  float* scale = (float*)(apack + APACK_DW);

  int n4 = (int)(XB_BYTES / 4);
  prep_x_kernel<<<(n4 + 255) / 256, 256, 0, stream>>>(x, (uint32_t*)xb, n4);
  prep_w_scale_kernel<<<K_, 64, 0, stream>>>(w, scale);
  prep_w_pack_kernel<<<(int)((APACK_DW + 255) / 256), 256, 0, stream>>>(w, apack);

  int blocks = N_ * H_ * (W_ / 16);
  bireal_gemm_kernel<<<blocks, 256, 0, stream>>>(xb, (const v8i*)apack, scale, out);
}
